// SutraV05PeriLN_34084860461132
// MI455X (gfx1250) — matrix operand order repaired and checked by reading, not compiled, not run on hardware
//
#include <hip/hip_runtime.h>
#include <hip/hip_bf16.h>
#include <math.h>

typedef __bf16 bf16_t;
typedef __attribute__((ext_vector_type(16))) __bf16 v16bf;
typedef __attribute__((ext_vector_type(8)))  __bf16 v8bf;
typedef __attribute__((ext_vector_type(8)))  float   v8f;

constexpr int DIMc = 768;
constexpr int FFc  = 3072;
constexpr int Sc   = 6;
constexpr int Tc   = 1024;
constexpr int Bc   = 2;
constexpr int Mc   = Bc * Tc;   // 2048 tokens
constexpr int Vc   = 32000;

// ---------------- device helpers ----------------

__device__ __forceinline__ float gelu_f(float x) {
    // jax.nn.gelu default (tanh approximation)
    float x3 = x * x * x;
    return 0.5f * x * (1.0f + tanhf(0.7978845608028654f * (x + 0.044715f * x3)));
}

__device__ __forceinline__ float softplus_f(float x) {
    return (x > 20.0f) ? x : log1pf(expf(x));
}

__device__ __forceinline__ float waveSum(float v) {
    #pragma unroll
    for (int o = 16; o > 0; o >>= 1) v += __shfl_down(v, o, 32);
    return v;
}

__device__ __forceinline__ float blockSumN(float v, float* sred, int nthreads) {
    int tid = threadIdx.x;
    sred[tid] = v;
    __syncthreads();
    for (int s = nthreads >> 1; s > 0; s >>= 1) {
        if (tid < s) sred[tid] += sred[tid + s];
        __syncthreads();
    }
    float r = sred[0];
    __syncthreads();
    return r;
}

// CDNA5 async global->LDS copy (ASYNCcnt-tracked; bypasses VGPRs).
// lds_off = low 32 bits of the generic shared pointer == LDS byte offset.
__device__ __forceinline__ void async_copy_b128(unsigned lds_off, const void* gptr) {
    asm volatile("global_load_async_to_lds_b128 %0, %1, off"
                 :: "v"(lds_off), "v"(gptr)
                 : "memory");
}
__device__ __forceinline__ void wait_async0() {
    asm volatile("s_wait_asynccnt 0" ::: "memory");
}

union BfFrag { v16bf v; v8bf h[2]; };

// ---------------- WMMA GEMM ----------------
// C[z] = act( A[z] (MxK bf16) * B[z] (KxN f32->bf16) * scale + bias[z] )
// Block: 256 thr = 8 waves. Block tile 128x64, wave tile 32x32 (4x 16x16 wmma).
// B_T==true: B stored as (N,K) row-major (used for emb_w^T logits GEMM).
template<int ACT, bool OUT_BF16, bool B_T, bool HAS_BIAS>
__global__ __launch_bounds__(256)
void gemm_bf16_wmma(const bf16_t* __restrict__ A, const float* __restrict__ Bw,
                    const float* __restrict__ bias, void* __restrict__ Cv,
                    int M, int N, int K,
                    long long aZ, long long bZ, long long cZ, long long biasZ,
                    float scale)
{
    __shared__ bf16_t sA[128][40];   // 40 pad keeps 16B alignment (80B rows)
    __shared__ bf16_t sB[64][40];    // transposed tile: [n][k]

    const int z = blockIdx.z;
    const bf16_t* __restrict__ Ab = A + (long long)z * aZ;
    const float*  __restrict__ Bb = Bw + (long long)z * bZ;
    const float*  __restrict__ biasb = HAS_BIAS ? (bias + (long long)z * biasZ) : nullptr;

    const int bm = blockIdx.y * 128;
    const int bn = blockIdx.x * 64;
    const int tid  = threadIdx.x;
    const int lane = tid & 31;
    const int wave = tid >> 5;
    const int wm = (wave & 3) * 32;   // 4 waves along M
    const int wn = (wave >> 2) * 32;  // 2 waves along N

    v8f acc[2][2] = {};

    // Per-thread A-tile staging addresses (fixed across K loop)
    const int arow = tid >> 1;
    const int aoff = (tid & 1) * 16;
    const unsigned aLds = (unsigned)(size_t)&sA[arow][aoff];

    for (int k0 = 0; k0 < K; k0 += 32) {
        // ---- stage A tile: 128 x 32 bf16 via async global->LDS (no VGPR round trip) ----
        {
            const bf16_t* src = Ab + (long long)(bm + arow) * K + (k0 + aoff);
            async_copy_b128(aLds,      src);
            async_copy_b128(aLds + 16, src + 8);
            if (k0 + 32 < K) __builtin_prefetch(src + 32, 0, 1);
        }
        // ---- stage B tile: 32(K) x 64(N) f32 -> bf16, transposed into sB[n][k] ----
        // (needs VALU conversion, overlaps with the async A copy above)
        if (!B_T) {
            int kr = tid >> 3;            // 0..31
            int nc = (tid & 7) * 8;       // 0..56
            const float* src = Bb + (long long)(k0 + kr) * N + (bn + nc);
            if (k0 + 32 < K) __builtin_prefetch(src + 32LL * N, 0, 1);
            #pragma unroll
            for (int j = 0; j < 8; ++j) sB[nc + j][kr] = (bf16_t)src[j];
        } else {
            int n    = tid >> 2;          // 0..63
            int koff = (tid & 3) * 8;     // 0..24
            const float* src = Bb + (long long)(bn + n) * K + (k0 + koff);
            if (k0 + 32 < K) __builtin_prefetch(src + 32, 0, 1);
            #pragma unroll
            for (int j = 0; j < 8; ++j) sB[n][koff + j] = (bf16_t)src[j];
        }
        wait_async0();       // this wave's async A-tile writes landed in LDS
        __syncthreads();     // all waves' A/B staging visible

        // ---- fragments (contiguous k = 16*(lane/16) + s, same order for A and B) ----
        const int mrow = wm + (lane & 15);
        const int kbA  = (lane >> 4) * 16;
        const int ncol = wn + (lane & 15);
        const int kbB  = (lane >> 4) * 16;

        BfFrag af[2], bfg[2];
        #pragma unroll
        for (int i = 0; i < 2; ++i) {
            af[i].h[0] = *(const v8bf*)&sA[mrow + i * 16][kbA];
            af[i].h[1] = *(const v8bf*)&sA[mrow + i * 16][kbA + 8];
        }
        #pragma unroll
        for (int j = 0; j < 2; ++j) {
            bfg[j].h[0] = *(const v8bf*)&sB[ncol + j * 16][kbB];
            bfg[j].h[1] = *(const v8bf*)&sB[ncol + j * 16][kbB + 8];
        }
        #pragma unroll
        for (int i = 0; i < 2; ++i)
            #pragma unroll
            for (int j = 0; j < 2; ++j)
                acc[i][j] = __builtin_amdgcn_wmma_f32_16x16x32_bf16(
                    false, af[i].v, false, bfg[j].v, (short)0, acc[i][j], false, false);
        __syncthreads();
    }

    // ---- epilogue: scale + bias + activation + store (bias hoisted per lane) ----
    float bj[2] = {0.f, 0.f};
    if (HAS_BIAS) {
        bj[0] = biasb[bn + wn + (lane & 15)];
        bj[1] = biasb[bn + wn + 16 + (lane & 15)];
    }
    #pragma unroll
    for (int i = 0; i < 2; ++i) {
        #pragma unroll
        for (int j = 0; j < 2; ++j) {
            const int gn = bn + wn + j * 16 + (lane & 15);
            #pragma unroll
            for (int e = 0; e < 8; ++e) {
                const int gm = bm + wm + i * 16 + ((lane >> 4) ? (e + 8) : e);
                float val = acc[i][j][e] * scale + bj[j];
                if (ACT == 1) val = gelu_f(val);
                else if (ACT == 2) val = softplus_f(val) + 0.1f;
                long long idx = (long long)z * cZ + (long long)gm * N + gn;
                if (OUT_BF16) ((bf16_t*)Cv)[idx] = (bf16_t)val;
                else          ((float*)Cv)[idx]  = val;
            }
        }
    }
}

// ---------------- embedding + position ----------------
__global__ __launch_bounds__(256)
void embed_kernel(const int* __restrict__ x, const float* __restrict__ emb,
                  const float* __restrict__ pos, bf16_t* __restrict__ hOut)
{
    int token = blockIdx.x;
    int t = token % Tc;
    long long erow = (long long)x[token] * DIMc;
    #pragma unroll
    for (int i = 0; i < 3; ++i) {
        int d = threadIdx.x + i * 256;
        hOut[(long long)token * DIMc + d] = (bf16_t)(emb[erow + d] + pos[(long long)t * DIMc + d]);
    }
}

// ---------------- f32 -> bf16 convert ----------------
__global__ __launch_bounds__(256)
void cvt_bf16_kernel(const float* __restrict__ s, bf16_t* __restrict__ d, long long n)
{
    long long i = (long long)blockIdx.x * 256 + threadIdx.x;
    if (i < n) d[i] = (bf16_t)s[i];
}

// ---------------- pi init ----------------
__global__ __launch_bounds__(256)
void pi_init_kernel(float* __restrict__ pi)
{
    int i = blockIdx.x * 256 + threadIdx.x;
    if (i < Mc * Sc) pi[i] = ((i % Sc) == 2) ? 1.0f : 0.0f;
}

// ---------------- LayerNorm (f32 in, optional f32/bf16 out) ----------------
__global__ __launch_bounds__(256)
void ln_kernel(const float* __restrict__ src, float* __restrict__ dstF,
               bf16_t* __restrict__ dstB, const float* __restrict__ g,
               const float* __restrict__ b, int which)
{
    __shared__ float sred[256];
    int token = blockIdx.x;
    const float* xr = src + (long long)token * DIMc;
    float v[3], s1 = 0.f, s2 = 0.f;
    #pragma unroll
    for (int i = 0; i < 3; ++i) {
        v[i] = xr[threadIdx.x + i * 256];
        s1 += v[i]; s2 += v[i] * v[i];
    }
    float mean = blockSumN(s1, sred, 256) * (1.0f / DIMc);
    float var  = blockSumN(s2, sred, 256) * (1.0f / DIMc) - mean * mean;
    float rstd = rsqrtf(var + 1e-5f);
    #pragma unroll
    for (int i = 0; i < 3; ++i) {
        int d = threadIdx.x + i * 256;
        float o = (v[i] - mean) * rstd * g[which * DIMc + d] + b[which * DIMc + d];
        if (dstF) dstF[(long long)token * DIMc + d] = o;
        if (dstB) dstB[(long long)token * DIMc + d] = (bf16_t)o;
    }
}

// ---------------- routing: kz -> K softmax -> pi_ev; ev softmax; top-2 pi ----------------
__global__ __launch_bounds__(256)
void route_kernel(const bf16_t* __restrict__ g1, const float* __restrict__ w2,
                  const float* __restrict__ b2, const bf16_t* __restrict__ mun,
                  const float* __restrict__ evw, const float* __restrict__ evb,
                  const float* __restrict__ piCur, float* __restrict__ piNext)
{
    __shared__ float swave[8 * 42];
    __shared__ float tot[42];
    int token = blockIdx.x;
    int tid = threadIdx.x, lane = tid & 31, wave = tid >> 5;

    float accK[36];
    #pragma unroll
    for (int j = 0; j < 36; ++j) accK[j] = 0.f;
    const bf16_t* g = g1 + (long long)token * (2 * DIMc);
    for (int k = tid; k < 2 * DIMc; k += 256) {
        float a = (float)g[k];
        const float* wr = w2 + (long long)k * 36;
        #pragma unroll
        for (int j = 0; j < 36; ++j) accK[j] += a * wr[j];
    }
    float accE[6];
    #pragma unroll
    for (int j = 0; j < 6; ++j) accE[j] = 0.f;
    const bf16_t* mn = mun + (long long)token * DIMc;
    for (int k = tid; k < DIMc; k += 256) {
        float a = (float)mn[k];
        const float* wr = evw + (long long)k * Sc;
        #pragma unroll
        for (int j = 0; j < 6; ++j) accE[j] += a * wr[j];
    }
    #pragma unroll
    for (int j = 0; j < 36; ++j) {
        float r = waveSum(accK[j]);
        if (lane == 0) swave[wave * 42 + j] = r;
    }
    #pragma unroll
    for (int j = 0; j < 6; ++j) {
        float r = waveSum(accE[j]);
        if (lane == 0) swave[wave * 42 + 36 + j] = r;
    }
    __syncthreads();
    if (tid < 42) {
        float s = 0.f;
        #pragma unroll
        for (int w = 0; w < 8; ++w) s += swave[w * 42 + tid];
        tot[tid] = s;
    }
    __syncthreads();
    if (tid == 0) {
        float Kmat[6][6], piev[6], ev[6], pin[6];
        // K = softmax over rows of (kz + b2)
        for (int s = 0; s < 6; ++s) {
            float mx = -1e30f;
            for (int u = 0; u < 6; ++u) {
                Kmat[s][u] = tot[s * 6 + u] + b2[s * 6 + u];
                mx = fmaxf(mx, Kmat[s][u]);
            }
            float sm = 0.f;
            for (int u = 0; u < 6; ++u) { Kmat[s][u] = expf(Kmat[s][u] - mx); sm += Kmat[s][u]; }
            for (int u = 0; u < 6; ++u) Kmat[s][u] /= sm;
        }
        for (int u = 0; u < 6; ++u) {
            float a = 0.f;
            for (int s = 0; s < 6; ++s) a += piCur[token * Sc + s] * Kmat[s][u];
            piev[u] = a;
        }
        // ev softmax / 0.5
        float mx = -1e30f;
        for (int u = 0; u < 6; ++u) { ev[u] = (tot[36 + u] + evb[u]) * 2.0f; mx = fmaxf(mx, ev[u]); }
        float sm = 0.f;
        for (int u = 0; u < 6; ++u) { ev[u] = expf(ev[u] - mx); sm += ev[u]; }
        float total = 0.f;
        for (int u = 0; u < 6; ++u) { pin[u] = piev[u] * (ev[u] / sm); total += pin[u]; }
        float inv = 1.0f / fmaxf(total, 1e-8f);
        for (int u = 0; u < 6; ++u) pin[u] *= inv;
        // top-2 (first occurrence on ties, matching lax.top_k)
        int i1 = 0;
        for (int u = 1; u < 6; ++u) if (pin[u] > pin[i1]) i1 = u;
        int i2 = -1;
        for (int u = 0; u < 6; ++u) if (u != i1 && (i2 < 0 || pin[u] > pin[i2])) i2 = u;
        float s2s = fmaxf(pin[i1] + pin[i2], 1e-8f);
        for (int u = 0; u < 6; ++u)
            piNext[token * Sc + u] = (u == i1 || u == i2) ? pin[u] / s2s : 0.f;
    }
}

// ---------------- combine bank outputs with old pi, then LN(1) ----------------
__global__ __launch_bounds__(256)
void combine_kernel(const float* __restrict__ so_s, const float* __restrict__ piCur,
                    const float* __restrict__ g, const float* __restrict__ b,
                    float* __restrict__ soOut)
{
    __shared__ float sred[256];
    int token = blockIdx.x;
    float ppi[6];
    #pragma unroll
    for (int s = 0; s < 6; ++s) ppi[s] = piCur[token * Sc + s];
    float v[3], s1 = 0.f, s2 = 0.f;
    #pragma unroll
    for (int i = 0; i < 3; ++i) {
        int d = threadIdx.x + i * 256;
        float a = 0.f;
        #pragma unroll
        for (int s = 0; s < 6; ++s)
            a += ppi[s] * so_s[(long long)s * Mc * DIMc + (long long)token * DIMc + d];
        v[i] = a; s1 += a; s2 += a * a;
    }
    float mean = blockSumN(s1, sred, 256) * (1.0f / DIMc);
    float var  = blockSumN(s2, sred, 256) * (1.0f / DIMc) - mean * mean;
    float rstd = rsqrtf(var + 1e-5f);
    #pragma unroll
    for (int i = 0; i < 3; ++i) {
        int d = threadIdx.x + i * 256;
        soOut[(long long)token * DIMc + d] = (v[i] - mean) * rstd * g[1 * DIMc + d] + b[1 * DIMc + d];
    }
}

// ---------------- local attention over OFFSETS + gate + LN(3) ----------------
__global__ __launch_bounds__(128)
void attn_kernel(const bf16_t* __restrict__ q, const bf16_t* __restrict__ k,
                 const bf16_t* __restrict__ v, const float* __restrict__ piNext,
                 const float* __restrict__ g, const float* __restrict__ b,
                 bf16_t* __restrict__ msgOut)
{
    __shared__ float ssc[4], swt[4];
    __shared__ float smsg[DIMc];
    __shared__ float sred[128];
    const int offs[4] = {-2, -1, 1, 2};
    int token = blockIdx.x;
    int bb = token / Tc, t = token % Tc;
    int lane = threadIdx.x & 31, wave = threadIdx.x >> 5;

    int tp = t + offs[wave];
    bool valid = (tp >= 0) && (tp < Tc);
    float d = 0.f;
    if (valid) {
        const bf16_t* qp = q + (long long)token * DIMc;
        const bf16_t* kp = k + (long long)(bb * Tc + tp) * DIMc;
        for (int i = lane; i < DIMc; i += 32) d += (float)qp[i] * (float)kp[i];
    }
    d = waveSum(d);
    if (lane == 0) ssc[wave] = valid ? d * rsqrtf((float)DIMc) : -1e9f;
    __syncthreads();
    if (threadIdx.x == 0) {
        float mx = fmaxf(fmaxf(ssc[0], ssc[1]), fmaxf(ssc[2], ssc[3]));
        float sm = 0.f, e[4];
        for (int o = 0; o < 4; ++o) { e[o] = expf(ssc[o] - mx); sm += e[o]; }
        for (int o = 0; o < 4; ++o) swt[o] = e[o] / sm;
    }
    __syncthreads();
    float gate = piNext[token * Sc + 3];
    for (int i = threadIdx.x; i < DIMc; i += 128) {
        float m = 0.f;
        #pragma unroll
        for (int o = 0; o < 4; ++o) {
            int tpo = t + offs[o];
            if (tpo >= 0 && tpo < Tc)
                m += swt[o] * (float)v[(long long)(bb * Tc + tpo) * DIMc + i];
        }
        smsg[i] = m * gate;
    }
    __syncthreads();
    float s1 = 0.f, s2 = 0.f;
    for (int i = threadIdx.x; i < DIMc; i += 128) { float a = smsg[i]; s1 += a; s2 += a * a; }
    float mean = blockSumN(s1, sred, 128) * (1.0f / DIMc);
    float var  = blockSumN(s2, sred, 128) * (1.0f / DIMc) - mean * mean;
    float rstd = rsqrtf(var + 1e-5f);
    for (int i = threadIdx.x; i < DIMc; i += 128)
        msgOut[(long long)token * DIMc + i] =
            (bf16_t)((smsg[i] - mean) * rstd * g[3 * DIMc + i] + b[3 * DIMc + i]);
}

// ---------------- lambda/mu update + LN(5) + so residual ----------------
__global__ __launch_bounds__(256)
void update_kernel(float* __restrict__ muF, bf16_t* __restrict__ muB,
                   float* __restrict__ lam, const float* __restrict__ wl,
                   const float* __restrict__ mhat, const float* __restrict__ muwn,
                   const float* __restrict__ so, const float* __restrict__ piNext,
                   const float* __restrict__ g, const float* __restrict__ b)
{
    __shared__ float sred[256];
    int token = blockIdx.x;
    float gpi = piNext[token * Sc + 4];
    float v[3], s1 = 0.f, s2 = 0.f;
    #pragma unroll
    for (int i = 0; i < 3; ++i) {
        long long idx = (long long)token * DIMc + threadIdx.x + i * 256;
        float dl = gpi * softplus_f(wl[idx]);
        float lo = lam[idx];
        float ln2 = lo + dl;
        lam[idx] = ln2;
        float val = (lo * muwn[idx] + dl * mhat[idx]) / ln2;
        v[i] = val; s1 += val; s2 += val * val;
    }
    float mean = blockSumN(s1, sred, 256) * (1.0f / DIMc);
    float var  = blockSumN(s2, sred, 256) * (1.0f / DIMc) - mean * mean;
    float rstd = rsqrtf(var + 1e-5f);
    #pragma unroll
    for (int i = 0; i < 3; ++i) {
        int d = threadIdx.x + i * 256;
        long long idx = (long long)token * DIMc + d;
        float o = (v[i] - mean) * rstd * g[5 * DIMc + d] + b[5 * DIMc + d] + 0.1f * so[idx];
        muF[idx] = o;
        muB[idx] = (bf16_t)o;
    }
}

// ---------------- host launcher ----------------
extern "C" void kernel_launch(void* const* d_in, const int* in_sizes, int n_in,
                              void* d_out, int out_size, void* d_ws, size_t ws_size,
                              hipStream_t stream)
{
    (void)in_sizes; (void)n_in; (void)out_size; (void)ws_size;

    const int*   x        = (const int*)  d_in[0];
    const float* emb_w    = (const float*)d_in[1];
    const float* pos_w    = (const float*)d_in[2];
    const float* mu_w     = (const float*)d_in[3];
    const float* mu_b     = (const float*)d_in[4];
    const float* lam_w    = (const float*)d_in[5];
    const float* lam_b    = (const float*)d_in[6];
    const float* tr_w1    = (const float*)d_in[7];
    const float* tr_b1    = (const float*)d_in[8];
    const float* tr_w2    = (const float*)d_in[9];
    const float* tr_b2    = (const float*)d_in[10];
    const float* bank_w1  = (const float*)d_in[11];
    const float* bank_b1  = (const float*)d_in[12];
    const float* bank_w2  = (const float*)d_in[13];
    const float* bank_b2  = (const float*)d_in[14];
    const float* ev_w     = (const float*)d_in[15];
    const float* ev_b     = (const float*)d_in[16];
    const float* rt_q     = (const float*)d_in[17];
    const float* rt_k     = (const float*)d_in[18];
    const float* rt_v     = (const float*)d_in[19];
    const float* wr_lam_w = (const float*)d_in[20];
    const float* wr_lam_b = (const float*)d_in[21];
    const float* wr_mu_w  = (const float*)d_in[22];
    const float* wr_mu_b  = (const float*)d_in[23];
    const float* ln_g     = (const float*)d_in[24];
    const float* ln_b     = (const float*)d_in[25];
    float* out = (float*)d_out;

    char* p = (char*)d_ws;
    auto alloc = [&](size_t bytes) -> char* {
        char* r = p; p += (bytes + 255) & ~(size_t)255; return r;
    };
    const size_t MD = (size_t)Mc * DIMc;
    bf16_t* h_bf    = (bf16_t*)alloc(MD * 2);
    float*  mu_f    = (float*) alloc(MD * 4);
    bf16_t* mu_bf   = (bf16_t*)alloc(MD * 2);
    float*  lam_f   = (float*) alloc(MD * 4);
    bf16_t* mun_bf  = (bf16_t*)alloc(MD * 2);
    bf16_t* g1_bf   = (bf16_t*)alloc((size_t)Mc * 2 * DIMc * 2);
    bf16_t* hb_bf   = (bf16_t*)alloc((size_t)Sc * Mc * FFc * 2);
    float*  sos_f   = (float*) alloc((size_t)Sc * MD * 4);
    float*  so_f    = (float*) alloc(MD * 4);
    float*  piA     = (float*) alloc((size_t)Mc * Sc * 4);
    float*  piB     = (float*) alloc((size_t)Mc * Sc * 4);
    bf16_t* mur_bf  = (bf16_t*)alloc(MD * 2);
    bf16_t* q_bf    = (bf16_t*)alloc(MD * 2);
    bf16_t* k_bf    = (bf16_t*)alloc(MD * 2);
    bf16_t* v_bf    = (bf16_t*)alloc(MD * 2);
    bf16_t* msg_bf  = (bf16_t*)alloc(MD * 2);
    float*  wl_f    = (float*) alloc(MD * 4);
    float*  mhat_f  = (float*) alloc(MD * 4);
    float*  muwn_f  = (float*) alloc(MD * 4);
    bf16_t* fin_bf  = (bf16_t*)alloc(MD * 2);

    const dim3 blk(256);
    const dim3 gD(DIMc / 64, Mc / 128, 1);       // N = 768
    const dim3 g2D(2 * DIMc / 64, Mc / 128, 1);  // N = 1536
    const dim3 gFF(FFc / 64, Mc / 128, Sc);      // N = 3072, z = S
    const dim3 gDS(DIMc / 64, Mc / 128, Sc);     // N = 768,  z = S
    const dim3 gV(Vc / 64, Mc / 128, 1);         // N = 32000

    // h = emb[x] + pos
    embed_kernel<<<Mc, blk, 0, stream>>>(x, emb_w, pos_w, h_bf);
    // mu = h @ mu_w + mu_b
    gemm_bf16_wmma<0, false, false, true><<<gD, blk, 0, stream>>>(
        h_bf, mu_w, mu_b, mu_f, Mc, DIMc, DIMc, 0, 0, 0, 0, 1.0f);
    cvt_bf16_kernel<<<(int)((MD + 255) / 256), blk, 0, stream>>>(mu_f, mu_bf, (long long)MD);
    // lam = softplus(h @ lam_w + lam_b) + 0.1
    gemm_bf16_wmma<2, false, false, true><<<gD, blk, 0, stream>>>(
        h_bf, lam_w, lam_b, lam_f, Mc, DIMc, DIMc, 0, 0, 0, 0, 1.0f);
    pi_init_kernel<<<(Mc * Sc + 255) / 256, blk, 0, stream>>>(piA);

    for (int step = 0; step < 3; ++step) {
        float* piCur  = (step & 1) ? piB : piA;
        float* piNext = (step & 1) ? piA : piB;

        // mu_n = ln(mu, 0)
        ln_kernel<<<Mc, blk, 0, stream>>>(mu_f, nullptr, mun_bf, ln_g, ln_b, 0);
        // g1 = gelu(mu @ tr_w1 + tr_b1)
        gemm_bf16_wmma<1, true, false, true><<<g2D, blk, 0, stream>>>(
            mu_bf, tr_w1, tr_b1, g1_bf, Mc, 2 * DIMc, DIMc, 0, 0, 0, 0, 1.0f);
        // routing: K softmax, pi_ev, ev softmax, top-2 -> piNext
        route_kernel<<<Mc, blk, 0, stream>>>(g1_bf, tr_w2, tr_b2, mun_bf, ev_w, ev_b,
                                             piCur, piNext);
        // hb[s] = gelu(mu_n @ bank_w1[s] + bank_b1[s])
        gemm_bf16_wmma<1, true, false, true><<<gFF, blk, 0, stream>>>(
            mun_bf, bank_w1, bank_b1, hb_bf, Mc, FFc, DIMc,
            0, (long long)DIMc * FFc, (long long)Mc * FFc, FFc, 1.0f);
        // so_s[s] = hb[s] @ bank_w2[s] + bank_b2[s]
        gemm_bf16_wmma<0, false, false, true><<<gDS, blk, 0, stream>>>(
            hb_bf, bank_w2, bank_b2, sos_f, Mc, DIMc, FFc,
            (long long)Mc * FFc, (long long)FFc * DIMc, (long long)Mc * DIMc, DIMc, 1.0f);
        // so = ln1( sum_s piCur_s * so_s )
        combine_kernel<<<Mc, blk, 0, stream>>>(sos_f, piCur, ln_g, ln_b, so_f);
        // mu_r = ln(mu, 2); q/k/v
        ln_kernel<<<Mc, blk, 0, stream>>>(mu_f, nullptr, mur_bf, ln_g, ln_b, 2);
        gemm_bf16_wmma<0, true, false, false><<<gD, blk, 0, stream>>>(
            mur_bf, rt_q, nullptr, q_bf, Mc, DIMc, DIMc, 0, 0, 0, 0, 1.0f);
        gemm_bf16_wmma<0, true, false, false><<<gD, blk, 0, stream>>>(
            mur_bf, rt_k, nullptr, k_bf, Mc, DIMc, DIMc, 0, 0, 0, 0, 1.0f);
        gemm_bf16_wmma<0, true, false, false><<<gD, blk, 0, stream>>>(
            mur_bf, rt_v, nullptr, v_bf, Mc, DIMc, DIMc, 0, 0, 0, 0, 1.0f);
        // messages = ln3( attn * piNext[:,3] )
        attn_kernel<<<Mc, dim3(128), 0, stream>>>(q_bf, k_bf, v_bf, piNext, ln_g, ln_b, msg_bf);
        // wl = messages @ wr_lam_w + b ; m_hat = messages @ wr_mu_w + b
        gemm_bf16_wmma<0, false, false, true><<<gD, blk, 0, stream>>>(
            msg_bf, wr_lam_w, wr_lam_b, wl_f, Mc, DIMc, DIMc, 0, 0, 0, 0, 1.0f);
        gemm_bf16_wmma<0, false, false, true><<<gD, blk, 0, stream>>>(
            msg_bf, wr_mu_w, wr_mu_b, mhat_f, Mc, DIMc, DIMc, 0, 0, 0, 0, 1.0f);
        // mu_w_n = ln(mu, 4)
        ln_kernel<<<Mc, blk, 0, stream>>>(mu_f, muwn_f, nullptr, ln_g, ln_b, 4);
        // lam/mu update, ln5, + 0.1*so
        update_kernel<<<Mc, blk, 0, stream>>>(mu_f, mu_bf, lam_f, wl_f, mhat_f,
                                              muwn_f, so_f, piNext, ln_g, ln_b);
    }

    // final = ln(mu, 6); logits = final @ emb_w.T / sqrt(D)
    ln_kernel<<<Mc, blk, 0, stream>>>(mu_f, nullptr, fin_bf, ln_g, ln_b, 6);
    gemm_bf16_wmma<0, false, true, false><<<gV, blk, 0, stream>>>(
        fin_bf, emb_w, nullptr, out, Mc, Vc, DIMc, 0, 0, 0, 0,
        1.0f / sqrtf((float)DIMc));
}
